// DWSpiralDeblock_10634339025473
// MI455X (gfx1250) — matrix operand order repaired and checked by reading, not compiled, not run on hardware
//
#include <hip/hip_runtime.h>

// MI455X / gfx1250, wave32. WMMA f16 path (codegen-confirmed builtin).
//
// Pipeline:
//   pool_kernel : pooled[b][v][c] = sum_k x[b, trans_col[row_map[v,k]], c] * trans_value[row_map[v,k]]
//                 -> d_ws (needs 8*16384*64*4 = 32 MB scratch)
//   fused_dw_pw : dw (depthwise, f32, -> f16 in LDS) then (16x64)x(64x32) tile GEMM via
//                 v_wmma_f32_16x16x32_f16 (two K=32 chunks x two N=16 halves), +bp, relu.

typedef __attribute__((ext_vector_type(16))) _Float16 v16h;
typedef __attribute__((ext_vector_type(8)))  float    v8f;

#define B_    8
#define VIN_  4096
#define CIN_  64
#define VOUT_ 16384
#define S_    9
#define COUT_ 32

union HFrag { v16h v; _Float16 e[16]; };

// ---------------- Kernel 1: gather + scale + 3-way pool --------------------
__global__ __launch_bounds__(256) void pool_kernel(
    const float* __restrict__ x,
    const int*   __restrict__ trans_col,
    const float* __restrict__ trans_value,
    const int*   __restrict__ row_map,
    float*       __restrict__ pooled)
{
    const int gid  = blockIdx.x * blockDim.x + threadIdx.x;
    const int wave = gid >> 5;               // one wave per (b,v) row
    const int lane = gid & 31;               // lane owns 2 channels (float2)
    if (wave >= B_ * VOUT_) return;
    const int b = wave >> 14;                // / VOUT_
    const int v = wave & (VOUT_ - 1);

    const float2* xb = (const float2*)(x + (size_t)b * VIN_ * CIN_);
    float2 acc = make_float2(0.f, 0.f);
#pragma unroll
    for (int k = 0; k < 3; ++k) {
        const int   e   = row_map[v * 3 + k];
        const int   col = trans_col[e];
        const float val = trans_value[e];
        const float2 xv = xb[col * (CIN_ / 2) + lane];
        acc.x = fmaf(val, xv.x, acc.x);
        acc.y = fmaf(val, xv.y, acc.y);
    }
    ((float2*)pooled)[(size_t)wave * (CIN_ / 2) + lane] = acc;
}

// -------- Kernel 2: fused depthwise + pointwise (WMMA) + bias + relu -------
// 128 threads = 4 waves; each wave owns one 16-row tile of the flattened
// (b*VOUT+v) dimension.  VOUT divisible by 16 -> tiles never straddle b.
__global__ __launch_bounds__(128) void fused_dw_pw_kernel(
    const float* __restrict__ pooled,
    const int*   __restrict__ indices,
    const float* __restrict__ Wd,     // (C_IN=64, S=9) row-major
    const float* __restrict__ bd,     // (64)
    const float* __restrict__ Wp,     // (C_OUT=32, C_IN=64) row-major
    const float* __restrict__ bp,     // (32)
    float*       __restrict__ out)    // (B, VOUT, 32)
{
    __shared__ _Float16 dwS[4][16][CIN_];    // 8 KB: f16 dw tiles, one per wave

    const int w    = threadIdx.x >> 5;
    const int lane = threadIdx.x & 31;
    const int tile = blockIdx.x * 4 + w;
    const int row0 = tile * 16;              // flattened row base (b*VOUT + v)
    const int b    = row0 >> 14;

    // ---- Stage 1: depthwise into LDS (f32 math, f16 store) ----
    float wd0[S_], wd1[S_];
#pragma unroll
    for (int s = 0; s < S_; ++s) {
        wd0[s] = Wd[lane * S_ + s];
        wd1[s] = Wd[(lane + 32) * S_ + s];
    }
    const float bd0 = bd[lane];
    const float bd1 = bd[lane + 32];

    for (int m = 0; m < 16; ++m) {
        const int row = row0 + m;
        const int v   = row & (VOUT_ - 1);
        float a0 = bd0, a1 = bd1;
#pragma unroll
        for (int s = 0; s < S_; ++s) {
            const int idx = indices[v * S_ + s];
            const float* pr = pooled + ((size_t)b * VOUT_ + idx) * CIN_;
            a0 = fmaf(pr[lane],      wd0[s], a0);
            a1 = fmaf(pr[lane + 32], wd1[s], a1);
        }
        dwS[w][m][lane]      = (_Float16)a0;
        dwS[w][m][lane + 32] = (_Float16)a1;
    }
    __syncthreads();

    // ---- Stage 2: (16x64) x (64x32) via v_wmma_f32_16x16x32_f16 ----
    const int mm    = lane & 15;
    const int kbase = (lane < 16) ? 0 : 8;   // ISA 16-bit A 16x32 layout

    HFrag afr[2];                            // K chunks 0..31 and 32..63
#pragma unroll
    for (int kk = 0; kk < 2; ++kk) {
#pragma unroll
        for (int h = 0; h < 16; ++h) {
            const int j = h >> 1, i = h & 1;
            const int k = (j < 4) ? (kbase + 2 * j + i)
                                  : (16 + kbase + 2 * (j - 4) + i);
            afr[kk].e[h] = dwS[w][mm][kk * 32 + k];
        }
    }

    const int koff = (lane < 16) ? 0 : 8;    // same k order as A
#pragma unroll
    for (int nh = 0; nh < 2; ++nh) {
        const int n = nh * 16 + mm;          // output column this lane holds
        HFrag bfr[2];
#pragma unroll
        for (int kk = 0; kk < 2; ++kk) {
#pragma unroll
            for (int h = 0; h < 16; ++h)
                bfr[kk].e[h] = (_Float16)Wp[n * CIN_ + kk * 32 + ((h < 8) ? (koff + h) : (16 + koff + h - 8))];
        }

        v8f acc = {};
        acc = __builtin_amdgcn_wmma_f32_16x16x32_f16(false, afr[0].v, false, bfr[0].v,
                                                     (short)0, acc, false, false);
        acc = __builtin_amdgcn_wmma_f32_16x16x32_f16(false, afr[1].v, false, bfr[1].v,
                                                     (short)0, acc, false, false);

        const float bpn  = bp[n];
        const int   mofs = (lane < 16) ? 0 : 8;   // C/D layout: VGPR r -> M = r (+8)
#pragma unroll
        for (int r = 0; r < 8; ++r) {
            const int   row = row0 + mofs + r;
            const float val = acc[r] + bpn;
            out[(size_t)row * COUT_ + n] = fmaxf(val, 0.f);
        }
    }
}

extern "C" void kernel_launch(void* const* d_in, const int* in_sizes, int n_in,
                              void* d_out, int out_size, void* d_ws, size_t ws_size,
                              hipStream_t stream) {
    const float* x           = (const float*)d_in[0];
    // d_in[1] = trans_row : unused by the reference computation
    const int*   trans_col   = (const int*)  d_in[2];
    const float* trans_value = (const float*)d_in[3];
    const int*   row_map     = (const int*)  d_in[4];
    const int*   indices     = (const int*)  d_in[5];
    const float* Wd          = (const float*)d_in[6];
    const float* bd          = (const float*)d_in[7];
    const float* Wp          = (const float*)d_in[8];
    const float* bp          = (const float*)d_in[9];
    float*       out         = (float*)d_out;

    float* pooled = (float*)d_ws;   // 8*16384*64 f32 = 32 MB of scratch

    // Kernel 1: one wave per (b,v) row -> 131072 waves, 8 waves/block
    pool_kernel<<<(B_ * VOUT_) / 8, 256, 0, stream>>>(
        x, trans_col, trans_value, row_map, pooled);

    // Kernel 2: 8192 tiles of 16 rows, 4 waves/block
    fused_dw_pw_kernel<<<(B_ * VOUT_ / 16) / 4, 128, 0, stream>>>(
        pooled, indices, Wd, bd, Wp, bp, out);
}
